// Transformer_33612414058612
// MI455X (gfx1250) — matrix operand order repaired and checked by reading, not compiled, not run on hardware
//
#include <hip/hip_runtime.h>

// ---------------------------------------------------------------------------
// Transformer encoder-decoder forward, CDNA5 (gfx1250), bf16 WMMA GEMM core.
// ---------------------------------------------------------------------------

#define DD    1024   // model dim
#define HH    16     // heads
#define DHH   64     // head dim
#define HIDD  4096   // ffn hidden
#define NBB   4      // blocks
#define BB    2      // batch
#define TT    1024   // seq len
#define BT    (BB * TT)
#define VTT   32000  // target vocab
#define EPSF  1e-5f

typedef __attribute__((ext_vector_type(16))) __bf16       v16bf;
typedef __attribute__((ext_vector_type(8)))  float        v8f;
typedef __attribute__((ext_vector_type(4)))  float        v4f;
typedef __attribute__((ext_vector_type(4)))  unsigned int v4u;
typedef __attribute__((ext_vector_type(2)))  unsigned int v2u;

union FragBF { v16bf v; v4u q[2]; };

__device__ __forceinline__ unsigned short cvt_bf16_rne(float f) {
  unsigned int u = __float_as_uint(f);
  u += 0x7fffu + ((u >> 16) & 1u);   // round-to-nearest-even
  return (unsigned short)(u >> 16);
}

// pack two fp32 -> two bf16 in one dword (lo = a, hi = b), RNE
__device__ __forceinline__ unsigned int pack_bf16(float a, float b) {
  unsigned int ua = __float_as_uint(a);
  unsigned int ub = __float_as_uint(b);
  ua += 0x7fffu + ((ua >> 16) & 1u);
  ub += 0x7fffu + ((ub >> 16) & 1u);
  return (ua >> 16) | (ub & 0xffff0000u);
}

// ---------------------------------------------------------------------------
// GEMM: C = alpha * (A @ B) + bias, optional ReLU.
//   A: [M,K] fp32, lda
//   B: bmode 0: [K,N] ldb        (normal)
//      bmode 1: [N,K] ldb        (transposed, for Q @ K^T)
//      bmode 2: [H][K][64] head-blocked (qw/kw/vw layout), ldb = K*64
//   C: [M,N] fp32, ldc
// Block = 256 threads (8 waves), tile 128x64; wave grid 4(m) x 2(n), each wave
// computes 32x32 via 2x2 WMMA tiles. K step 32, bf16 WMMA, f32 accumulate.
// Requires M%128==0, N%64==0, K%32==0 (true for every GEMM in this model).
// ---------------------------------------------------------------------------
__global__ __launch_bounds__(256)
void wmma_gemm_bf16(const float* __restrict__ A, const float* __restrict__ Bm,
                    const float* __restrict__ bias, float* __restrict__ C,
                    int K, int lda, int ldb, int ldc,
                    int bmode, float alpha, int relu) {
  __shared__ unsigned short As[128 * 32];  // [m][k]
  __shared__ unsigned short Bs[64 * 32];   // [n][k]

  const int tid  = threadIdx.x;
  const int wave = tid >> 5;
  const int lane = tid & 31;
  const int half = lane >> 4;
  const int l16  = lane & 15;
  const int wm   = wave >> 1;   // 0..3 -> 32-row strip
  const int wn   = wave & 1;    // 0..1 -> 32-col strip

  const int bm = blockIdx.y * 128;
  const int bn = blockIdx.x * 64;

  const v8f vzero = {0.f, 0.f, 0.f, 0.f, 0.f, 0.f, 0.f, 0.f};
  v8f acc[2][2];
  acc[0][0] = vzero; acc[0][1] = vzero; acc[1][0] = vzero; acc[1][1] = vzero;

  for (int k0 = 0; k0 < K; k0 += 32) {
    __syncthreads();
    // ---- A tile: 128x32, float4 loads, packed b64 LDS stores ----
#pragma unroll
    for (int i = 0; i < 4; ++i) {
      int idx4 = i * 256 + tid;            // 0..1023 float4 slots
      int r = idx4 >> 3, c4 = (idx4 & 7) << 2;
      const v4f f = *(const v4f*)(A + (long long)(bm + r) * lda + (k0 + c4));
      v2u p;
      p[0] = pack_bf16(f[0], f[1]);
      p[1] = pack_bf16(f[2], f[3]);
      *(v2u*)(&As[r * 32 + c4]) = p;
    }
    // ---- B tile: 64 n x 32 k, stored [n][k] ----
    if (bmode == 1) {                      // B is [N,K]: contiguous along k
#pragma unroll
      for (int i = 0; i < 2; ++i) {
        int idx4 = i * 256 + tid;          // 0..511
        int n = idx4 >> 3, c4 = (idx4 & 7) << 2;
        const v4f f = *(const v4f*)(Bm + (long long)(bn + n) * ldb + (k0 + c4));
        v2u p;
        p[0] = pack_bf16(f[0], f[1]);
        p[1] = pack_bf16(f[2], f[3]);
        *(v2u*)(&Bs[n * 32 + c4]) = p;
      }
    } else if (bmode == 0) {               // B is [K,N]: contiguous along n
#pragma unroll
      for (int i = 0; i < 2; ++i) {
        int idx4 = i * 256 + tid;
        int k = idx4 >> 4, n4 = (idx4 & 15) << 2;
        const v4f f = *(const v4f*)(Bm + (long long)(k0 + k) * ldb + (bn + n4));
#pragma unroll
        for (int j = 0; j < 4; ++j)
          Bs[(n4 + j) * 32 + k] = cvt_bf16_rne(f[j]);
      }
    } else {                               // bmode 2: [H][K][64], ldb = K*64
#pragma unroll
      for (int i = 0; i < 2; ++i) {
        int idx4 = i * 256 + tid;
        int k = idx4 >> 4, n4 = (idx4 & 15) << 2;
        int n = bn + n4;
        const v4f f = *(const v4f*)(Bm + (long long)(n >> 6) * ldb +
                                    (long long)(k0 + k) * 64 + (n & 63));
#pragma unroll
        for (int j = 0; j < 4; ++j)
          Bs[(n4 + j) * 32 + k] = cvt_bf16_rne(f[j]);
      }
    }
    if (k0 + 32 < K) {                     // hint next A tile -> global_prefetch_b8
      __builtin_prefetch(A + (long long)(bm + (tid >> 1)) * lda + (k0 + 32), 0, 1);
    }
    __syncthreads();

    // ---- fragments ----
    // A (ISA 16-bit A 16x32): lane row m = l16, k = 16*(j>=8) + 8*half + j%8
    FragBF fa[2], fb[2];
#pragma unroll
    for (int mt = 0; mt < 2; ++mt) {
      const v4u* pa = (const v4u*)(&As[(wm * 32 + mt * 16 + l16) * 32 + half * 8]);
      fa[mt].q[0] = pa[0];                 // k = half*8 .. +7
      fa[mt].q[1] = pa[2];                 // k = 16+half*8 .. +7
    }
    // B (ISA 16-bit B 32x16): lane col n = l16, k = 16*(j>=8) + 8*half + j%8
#pragma unroll
    for (int nt = 0; nt < 2; ++nt) {
      const v4u* pb = (const v4u*)(&Bs[(wn * 32 + nt * 16 + l16) * 32 + half * 8]);
      fb[nt].q[0] = pb[0];
      fb[nt].q[1] = pb[2];
    }
#pragma unroll
    for (int mt = 0; mt < 2; ++mt)
#pragma unroll
      for (int nt = 0; nt < 2; ++nt)
        acc[mt][nt] = __builtin_amdgcn_wmma_f32_16x16x32_bf16(
            false, fa[mt].v, false, fb[nt].v, (short)0, acc[mt][nt], false, false);
  }

  // ---- epilogue: D layout row = 8*half + r, col = l16 ----
#pragma unroll
  for (int mt = 0; mt < 2; ++mt)
#pragma unroll
    for (int nt = 0; nt < 2; ++nt) {
      int n = bn + wn * 32 + nt * 16 + l16;
      float bv = bias ? bias[n] : 0.0f;
#pragma unroll
      for (int r = 0; r < 8; ++r) {
        int m = bm + wm * 32 + mt * 16 + half * 8 + r;
        float o = acc[mt][nt][r] * alpha + bv;
        if (relu) o = fmaxf(o, 0.0f);
        C[(long long)m * ldc + n] = o;
      }
    }
}

// ---------------------------------------------------------------------------
// Row softmax over fixed row length TT with optional causal mask (row t keeps
// columns s <= t). One 256-thread block per row, 4 elements per thread.
// ---------------------------------------------------------------------------
__global__ __launch_bounds__(256)
void softmax_row_kernel(float* __restrict__ S, int causal) {
  const int t = blockIdx.x;
  float* row = S + (long long)t * TT;
  const int lim = causal ? (t + 1) : TT;
  __shared__ float sh[256];
  const int tid = threadIdx.x;

  float v[4];
  float m = -3.402823466e38f;
#pragma unroll
  for (int i = 0; i < 4; ++i) {
    int c = tid + i * 256;
    v[i] = row[c];
    if (c < lim) m = fmaxf(m, v[i]);
  }
  sh[tid] = m; __syncthreads();
  for (int s = 128; s > 0; s >>= 1) {
    if (tid < s) sh[tid] = fmaxf(sh[tid], sh[tid + s]);
    __syncthreads();
  }
  m = sh[0]; __syncthreads();

  float e[4]; float sum = 0.f;
#pragma unroll
  for (int i = 0; i < 4; ++i) {
    int c = tid + i * 256;
    e[i] = (c < lim) ? expf(v[i] - m) : 0.0f;
    sum += e[i];
  }
  sh[tid] = sum; __syncthreads();
  for (int s = 128; s > 0; s >>= 1) {
    if (tid < s) sh[tid] += sh[tid + s];
    __syncthreads();
  }
  float inv = 1.0f / sh[0];
#pragma unroll
  for (int i = 0; i < 4; ++i) {
    int c = tid + i * 256;
    row[c] = e[i] * inv;
  }
}

// ---------------------------------------------------------------------------
// out[row,:] = LayerNorm(x[row,:] + addv[row,:]) * g + b   (addv optional)
// Row length DD=1024; one 256-thread block per row. In-place safe.
// ---------------------------------------------------------------------------
__global__ __launch_bounds__(256)
void layernorm_kernel(const float* __restrict__ x, const float* __restrict__ addv,
                      const float* __restrict__ g, const float* __restrict__ b,
                      float* __restrict__ out) {
  const long long row = blockIdx.x;
  const float* xr = x + row * DD;
  const float* ar = addv ? (addv + row * DD) : nullptr;
  float* orow = out + row * DD;
  __shared__ float sh[256];
  const int tid = threadIdx.x;

  float v[4]; float s = 0.f;
#pragma unroll
  for (int i = 0; i < 4; ++i) {
    int c = tid + i * 256;
    v[i] = xr[c] + (ar ? ar[c] : 0.0f);
    s += v[i];
  }
  sh[tid] = s; __syncthreads();
  for (int t = 128; t > 0; t >>= 1) { if (tid < t) sh[tid] += sh[tid + t]; __syncthreads(); }
  float mean = sh[0] * (1.0f / DD); __syncthreads();

  float s2 = 0.f;
#pragma unroll
  for (int i = 0; i < 4; ++i) { float d = v[i] - mean; s2 += d * d; }
  sh[tid] = s2; __syncthreads();
  for (int t = 128; t > 0; t >>= 1) { if (tid < t) sh[tid] += sh[tid + t]; __syncthreads(); }
  float inv = rsqrtf(sh[0] * (1.0f / DD) + EPSF);

#pragma unroll
  for (int i = 0; i < 4; ++i) {
    int c = tid + i * 256;
    orow[c] = (v[i] - mean) * inv * g[c] + b[c];
  }
}

__global__ void add_inplace_kernel(float* __restrict__ a, const float* __restrict__ b,
                                   long long n) {
  long long i = (long long)blockIdx.x * blockDim.x + threadIdx.x;
  if (i < n) a[i] += b[i];
}

__global__ __launch_bounds__(256)
void embed_kernel(const int* __restrict__ ids, const float* __restrict__ tok,
                  const float* __restrict__ pos, float* __restrict__ out) {
  const int row = blockIdx.x;                // 0..BT-1, row = b*T + t
  const int t = row % TT;
  const long long id = ids[row];
  const float* tr = tok + id * DD;
  const float* pr = pos + (long long)t * DD;
  float* orow = out + (long long)row * DD;
  for (int c = threadIdx.x; c < DD; c += 256) orow[c] = tr[c] + pr[c];
}

// ---------------------------------------------------------------------------
// Host-side orchestration
// ---------------------------------------------------------------------------
struct BlkP {
  const float *qw, *qb, *kw, *kb, *vw, *vb, *pw, *pb;
  const float *ln1g, *ln1b, *ln2g, *ln2b, *w1, *b1, *w2, *b2;
  const float *mlng, *mlnb, *cqw, *cqb, *ckw, *ckb, *cvw, *cvb, *cpw, *cpb, *clng, *clnb;
};

static inline void launch_gemm(hipStream_t st, const float* A, const float* Bm,
                               const float* bias, float* C, int M, int N, int K,
                               int lda, int ldb, int ldc,
                               int bmode, float alpha, int relu) {
  dim3 grid(N / 64, M / 128, 1), block(256, 1, 1);
  wmma_gemm_bf16<<<grid, block, 0, st>>>(A, Bm, bias, C, K, lda, ldb, ldc,
                                         bmode, alpha, relu);
}

// multihead self-attention; out = proj(concat heads) + pb
static void self_attention(hipStream_t st, const float* x, const BlkP& p, bool masked,
                           float* q, float* k, float* v, float* ao, float* sc, float* out) {
  // fused over heads: Q/K/V = [BT, H*DH]; weights [H][D][DH] via bmode=2,
  // bias (H,DH) is contiguous 1024 floats -> plain bias[n].
  launch_gemm(st, x, p.qw, p.qb, q, BT, DD, DD, DD, DD * DHH, DD, 2, 1.0f, 0);
  launch_gemm(st, x, p.kw, p.kb, k, BT, DD, DD, DD, DD * DHH, DD, 2, 1.0f, 0);
  launch_gemm(st, x, p.vw, p.vb, v, BT, DD, DD, DD, DD * DHH, DD, 2, 1.0f, 0);
  const float scl = 0.125f;  // 1/sqrt(DH)
  for (int b = 0; b < BB; ++b) {
    for (int h = 0; h < HH; ++h) {
      const long long off = (long long)b * TT * DD + h * DHH;  // column slice h*64
      // scores = Q @ K^T * scl
      launch_gemm(st, q + off, k + off, nullptr, sc, TT, TT, DHH, DD, DD, TT,
                  1, scl, 0);
      softmax_row_kernel<<<dim3(TT), dim3(256), 0, st>>>(sc, masked ? 1 : 0);
      // O = P @ V, written into concat layout [B,T,H*DH]
      launch_gemm(st, sc, v + off, nullptr, ao + off, TT, DHH, TT, TT, DD, DD,
                  0, 1.0f, 0);
    }
  }
  launch_gemm(st, ao, p.pw, p.pb, out, BT, DD, DD, DD, DD, DD, 0, 1.0f, 0);
}

extern "C" void kernel_launch(void* const* d_in, const int* in_sizes, int n_in,
                              void* d_out, int out_size, void* d_ws, size_t ws_size,
                              hipStream_t stream) {
  (void)out_size; (void)ws_size;
  const int *src = nullptr, *tgt = nullptr;
  const float *enc_tok = nullptr, *enc_pos = nullptr, *dec_tok = nullptr,
              *dec_pos = nullptr, *lm_w = nullptr, *lm_b = nullptr;
  BlkP enc[NBB], dec[NBB];
  auto F = [&](int i) { return (const float*)d_in[i]; };

  const bool insertion = (n_in > 0 && in_sizes[0] == BT);  // src [B,T] first?
  if (insertion) {
    int i = 0;
    src = (const int*)d_in[i++]; tgt = (const int*)d_in[i++];
    enc_tok = F(i++); enc_pos = F(i++); dec_tok = F(i++); dec_pos = F(i++);
    for (int nb = 0; nb < NBB; ++nb) {
      BlkP& e = enc[nb];
      e.qw = F(i++); e.qb = F(i++); e.kw = F(i++); e.kb = F(i++);
      e.vw = F(i++); e.vb = F(i++); e.pw = F(i++); e.pb = F(i++);
      e.ln1g = F(i++); e.ln1b = F(i++); e.ln2g = F(i++); e.ln2b = F(i++);
      e.w1 = F(i++); e.b1 = F(i++); e.w2 = F(i++); e.b2 = F(i++);
    }
    for (int nb = 0; nb < NBB; ++nb) {
      BlkP& d = dec[nb];
      d.qw = F(i++); d.qb = F(i++); d.kw = F(i++); d.kb = F(i++);
      d.vw = F(i++); d.vb = F(i++); d.pw = F(i++); d.pb = F(i++);
      d.ln1g = F(i++); d.ln1b = F(i++); d.ln2g = F(i++); d.ln2b = F(i++);
      d.w1 = F(i++); d.b1 = F(i++); d.w2 = F(i++); d.b2 = F(i++);
      d.mlng = F(i++); d.mlnb = F(i++);
      d.cqw = F(i++); d.cqb = F(i++); d.ckw = F(i++); d.ckb = F(i++);
      d.cvw = F(i++); d.cvb = F(i++); d.cpw = F(i++); d.cpb = F(i++);
      d.clng = F(i++); d.clnb = F(i++);
    }
    lm_w = F(i++); lm_b = F(i++);
  } else {
    // jax pytree order: dict keys sorted alphabetically at every level
    int i = 0;
    for (int nb = 0; nb < NBB; ++nb) {
      BlkP& d = dec[nb];
      d.b1 = F(i++); d.b2 = F(i++); d.ckb = F(i++); d.ckw = F(i++);
      d.clnb = F(i++); d.clng = F(i++); d.cpb = F(i++); d.cpw = F(i++);
      d.cqb = F(i++); d.cqw = F(i++); d.cvb = F(i++); d.cvw = F(i++);
      d.kb = F(i++); d.kw = F(i++); d.ln1b = F(i++); d.ln1g = F(i++);
      d.ln2b = F(i++); d.ln2g = F(i++); d.mlnb = F(i++); d.mlng = F(i++);
      d.pb = F(i++); d.pw = F(i++); d.qb = F(i++); d.qw = F(i++);
      d.vb = F(i++); d.vw = F(i++); d.w1 = F(i++); d.w2 = F(i++);
    }
    dec_pos = F(i++); dec_tok = F(i++);
    for (int nb = 0; nb < NBB; ++nb) {
      BlkP& e = enc[nb];
      e.b1 = F(i++); e.b2 = F(i++); e.kb = F(i++); e.kw = F(i++);
      e.ln1b = F(i++); e.ln1g = F(i++); e.ln2b = F(i++); e.ln2g = F(i++);
      e.pb = F(i++); e.pw = F(i++); e.qb = F(i++); e.qw = F(i++);
      e.vb = F(i++); e.vw = F(i++); e.w1 = F(i++); e.w2 = F(i++);
    }
    enc_pos = F(i++); enc_tok = F(i++);
    lm_b = F(i++); lm_w = F(i++);
    src = (const int*)d_in[i++]; tgt = (const int*)d_in[i++];
  }

  // Workspace layout (fp32), ~100 MB total; every buffer fully written before read.
  float* ws = (float*)d_ws;
  float* enc_x = ws; ws += (long long)BT * DD;
  float* dec_x = ws; ws += (long long)BT * DD;
  float* t0    = ws; ws += (long long)BT * DD;
  float* t1    = ws; ws += (long long)BT * DD;
  float* qb_   = ws; ws += (long long)BT * DD;
  float* kb_   = ws; ws += (long long)BT * DD;
  float* vb_   = ws; ws += (long long)BT * DD;
  float* ao    = ws; ws += (long long)BT * DD;
  float* hid   = ws; ws += (long long)BT * HIDD;
  float* sc    = ws; ws += (long long)TT * TT;

  // ---- embeddings ----
  embed_kernel<<<dim3(BT), dim3(256), 0, stream>>>(src, enc_tok, enc_pos, enc_x);
  embed_kernel<<<dim3(BT), dim3(256), 0, stream>>>(tgt, dec_tok, dec_pos, dec_x);

  // ---- encoder ----
  for (int nb = 0; nb < NBB; ++nb) {
    const BlkP& p = enc[nb];
    self_attention(stream, enc_x, p, false, qb_, kb_, vb_, ao, sc, t0);
    layernorm_kernel<<<dim3(BT), dim3(256), 0, stream>>>(enc_x, t0, p.ln1g, p.ln1b, enc_x);
    launch_gemm(stream, enc_x, p.w1, p.b1, hid, BT, HIDD, DD, DD, HIDD, HIDD, 0, 1.0f, 1);
    launch_gemm(stream, hid, p.w2, p.b2, t0, BT, DD, HIDD, HIDD, DD, DD, 0, 1.0f, 0);
    layernorm_kernel<<<dim3(BT), dim3(256), 0, stream>>>(enc_x, t0, p.ln2g, p.ln2b, enc_x);
  }

  // ---- decoder ----
  for (int nb = 0; nb < NBB; ++nb) {
    const BlkP& p = dec[nb];
    // x1 = x + LN(mha(x, masked); mlng, mlnb)
    self_attention(stream, dec_x, p, true, qb_, kb_, vb_, ao, sc, t0);
    layernorm_kernel<<<dim3(BT), dim3(256), 0, stream>>>(t0, nullptr, p.mlng, p.mlnb, t0);
    add_inplace_kernel<<<dim3((BT * DD + 255) / 256), dim3(256), 0, stream>>>(
        dec_x, t0, (long long)BT * DD);
    // cross attention (full-D, no heads), scale 1/sqrt(D)
    launch_gemm(stream, dec_x, p.cqw, p.cqb, qb_, BT, DD, DD, DD, DD, DD, 0, 1.0f, 0);
    launch_gemm(stream, enc_x, p.ckw, p.ckb, kb_, BT, DD, DD, DD, DD, DD, 0, 1.0f, 0);
    launch_gemm(stream, enc_x, p.cvw, p.cvb, vb_, BT, DD, DD, DD, DD, DD, 0, 1.0f, 0);
    const float cscl = 0.03125f;  // 1/sqrt(1024)
    for (int b = 0; b < BB; ++b) {
      const long long off = (long long)b * TT * DD;
      launch_gemm(stream, qb_ + off, kb_ + off, nullptr, sc, TT, TT, DD, DD, DD, TT,
                  1, cscl, 0);
      softmax_row_kernel<<<dim3(TT), dim3(256), 0, stream>>>(sc, 0);
      launch_gemm(stream, sc, vb_ + off, nullptr, ao + off, TT, DD, TT, TT, DD, DD,
                  0, 1.0f, 0);
    }
    launch_gemm(stream, ao, p.cpw, p.cpb, t0, BT, DD, DD, DD, DD, DD, 0, 1.0f, 0);
    // x2 = LN(x1 + proj; clng, clnb)
    layernorm_kernel<<<dim3(BT), dim3(256), 0, stream>>>(dec_x, t0, p.clng, p.clnb, t1);
    // x3 = LN(x1 + x2; ln1)
    layernorm_kernel<<<dim3(BT), dim3(256), 0, stream>>>(dec_x, t1, p.ln1g, p.ln1b, dec_x);
    // x4 = LN(x3 + ffn(x3); ln2)
    launch_gemm(stream, dec_x, p.w1, p.b1, hid, BT, HIDD, DD, DD, HIDD, HIDD, 0, 1.0f, 1);
    launch_gemm(stream, hid, p.w2, p.b2, t0, BT, DD, HIDD, HIDD, DD, DD, 0, 1.0f, 0);
    layernorm_kernel<<<dim3(BT), dim3(256), 0, stream>>>(dec_x, t0, p.ln2g, p.ln2b, dec_x);
  }

  // ---- LM head: logits = dec_x @ lm_w + lm_b ----
  launch_gemm(stream, dec_x, lm_w, lm_b, (float*)d_out, BT, VTT, DD, DD, VTT, VTT,
              0, 1.0f, 0);
}
